// MultiheadAttention_75995151335505
// MI455X (gfx1250) — compile-verified, matrix operand order checked by reading, not run on hardware
//
#include <hip/hip_runtime.h>
#include <hip/hip_bf16.h>

// MI455X (gfx1250) multi-head attention:
//   q,k,v = hs @ W{q,k,v}.T + b    (f16 WMMA, f32 accumulate)
//   out   = softmax(q k^T / 8) v   (flash attention, f16 WMMA, f32 accumulate)
//
// Dims: B=4, S=2048, D=1024, H=16, HD=64.
// Workspace layout (bytes):
//   [0,        16MiB)   hidden states f16          (8192 x 1024)
//   [16MiB,    22MiB)   Wq|Wk|Wv f16               (3 x 1024 x 1024)
//   [22MiB,    38MiB)   Q f16  [B,H,S,64]
//   [38MiB,    54MiB)   K f16  [B,H,S,64]
//   [54MiB,    70MiB)   V f16  [B,H,64,S]  (transposed for P x V)

typedef __attribute__((ext_vector_type(16))) _Float16 v16h;
typedef __attribute__((ext_vector_type(8)))  _Float16 v8h;
typedef __attribute__((ext_vector_type(8)))  float    v8f;

#define SQ   2048
#define DIM  1024
#define HD   64
#define NH   16

// ---------------------------------------------------------------------------
// 16-bit A/B fragment load for v_wmma_f32_16x16x32_f16.
// ISA layout (16-bit A 16x32): lanes 0-15 hold K=0-7 (v0-3) and K=16-23 (v4-7),
// lanes 16-31 hold K=8-15 and K=24-31.  B columns use the same striping.
// `base` must already point at this lane's row/column (K-contiguous storage).
// ---------------------------------------------------------------------------
__device__ __forceinline__ v16h load_frag16(const _Float16* base, int lane) {
  const int ko = (lane >> 4) << 3;                  // 0 or 8
  const v8h lo = *(const v8h*)(base + ko);          // K = ko .. ko+7
  const v8h hi = *(const v8h*)(base + ko + 16);     // K = ko+16 .. ko+23
  v16h r;
#pragma unroll
  for (int i = 0; i < 8; ++i) { r[i] = lo[i]; r[i + 8] = hi[i]; }
  return r;
}

__device__ __forceinline__ v8f wmma16(v16h a, v16h b, v8f c) {
  return __builtin_amdgcn_wmma_f32_16x16x32_f16(
      /*neg_a=*/false, a, /*neg_b=*/false, b,
      /*c_mod=*/(short)0, c, /*reuse_a=*/false, /*reuse_b=*/false);
}

// ---------------------------------------------------------------------------
// fp32 -> fp16 conversion, 8 elements / thread, vectorized.
// ---------------------------------------------------------------------------
__global__ __launch_bounds__(256) void cvt_f32_to_f16(
    const float* __restrict__ s, _Float16* __restrict__ d, int n) {
  const int i = (blockIdx.x * blockDim.x + threadIdx.x) * 8;
  if (i >= n) return;
  const float4 x = *(const float4*)(s + i);
  const float4 y = *(const float4*)(s + i + 4);
  v8h o;
  o[0] = (_Float16)x.x; o[1] = (_Float16)x.y;
  o[2] = (_Float16)x.z; o[3] = (_Float16)x.w;
  o[4] = (_Float16)y.x; o[5] = (_Float16)y.y;
  o[6] = (_Float16)y.z; o[7] = (_Float16)y.w;
  *(v8h*)(d + i) = o;
}

// ---------------------------------------------------------------------------
// Fused QKV projection.  C[m,n] = hs[m,:] . W[n,:]  (+ bias[n]).
// 4 waves / block, each wave owns a 64x64 tile (4x4 WMMA tiles).
// grid = (M/128, N/128, 3); z selects q / k / v.
// Q,K stored [B,H,S,64]; V stored transposed [B,H,64,S].
// ---------------------------------------------------------------------------
__global__ __launch_bounds__(128) void qkv_gemm(
    const _Float16* __restrict__ A,   // hs f16, [8192,1024]
    const _Float16* __restrict__ Wf,  // Wq|Wk|Wv f16, each [1024,1024] row-major
    const float* __restrict__ bq, const float* __restrict__ bk,
    const float* __restrict__ bv,
    _Float16* __restrict__ qd, _Float16* __restrict__ kd,
    _Float16* __restrict__ vd) {
  const int z = blockIdx.z;
  const _Float16* W = Wf + (size_t)z * DIM * DIM;
  const float* bias = (z == 0) ? bq : (z == 1) ? bk : bv;
  _Float16* dst = (z == 0) ? qd : (z == 1) ? kd : vd;
  const bool vtrans = (z == 2);

  const int lane = threadIdx.x & 31;
  const int wv   = threadIdx.x >> 5;
  const int mw   = blockIdx.x * 128 + (wv >> 1) * 64;
  const int nw   = blockIdx.y * 128 + (wv & 1) * 64;
  const int mcol = lane & 15;
  const int mrow_off = (lane >> 4) << 3;            // 0 or 8 (C-layout row group)

  v8f c[4][4] = {};

#pragma unroll 1
  for (int kk = 0; kk < DIM; kk += 32) {
    v16h a[4], b[4];
#pragma unroll
    for (int i = 0; i < 4; ++i)
      a[i] = load_frag16(A + (size_t)(mw + i * 16 + mcol) * DIM + kk, lane);
#pragma unroll
    for (int j = 0; j < 4; ++j)
      b[j] = load_frag16(W + (size_t)(nw + j * 16 + mcol) * DIM + kk, lane);
#pragma unroll
    for (int i = 0; i < 4; ++i)
#pragma unroll
      for (int j = 0; j < 4; ++j)
        c[i][j] = wmma16(a[i], b[j], c[i][j]);
  }

  // bias + scatter to per-head layout
#pragma unroll
  for (int j = 0; j < 4; ++j) {
    const int n = nw + j * 16 + mcol;
    const float bvn = bias[n];
    const int h = n >> 6, d = n & 63;
#pragma unroll
    for (int i = 0; i < 4; ++i) {
#pragma unroll
      for (int r = 0; r < 8; ++r) {
        const int m  = mw + i * 16 + r + mrow_off;
        const int bb = m >> 11, s = m & (SQ - 1);
        const float val = c[i][j][r] + bvn;
        if (!vtrans)
          dst[(((size_t)(bb * NH + h)) * SQ + s) * HD + d] = (_Float16)val;
        else
          dst[(((size_t)(bb * NH + h)) * HD + d) * SQ + s] = (_Float16)val;
      }
    }
  }
}

// ---------------------------------------------------------------------------
// Flash attention: one wave per 16 query rows, online softmax over key
// chunks of 32.  8 waves / block -> 128 query rows per block.
// grid = (S/128, B*H).
// ---------------------------------------------------------------------------
__global__ __launch_bounds__(256) void attn_kernel(
    const _Float16* __restrict__ Q,   // [B,H,S,64]
    const _Float16* __restrict__ K,   // [B,H,S,64]
    const _Float16* __restrict__ Vt,  // [B,H,64,S]
    float* __restrict__ out) {        // [B,S,1024]
  __shared__ _Float16 plds[8][16 * 40];   // per-wave 16x32 P tile, padded rows

  const int lane = threadIdx.x & 31;
  const int wv   = threadIdx.x >> 5;
  const int bh   = blockIdx.y;                       // b*16 + h
  const int q0   = blockIdx.x * 128 + wv * 16;
  const int mcol = lane & 15;
  const int mrow_off = (lane >> 4) << 3;

  const _Float16* Qp = Q  + (size_t)bh * SQ * HD;
  const _Float16* Kp = K  + (size_t)bh * SQ * HD;
  const _Float16* Vp = Vt + (size_t)bh * HD * SQ;

  // Q A-fragments: rows q0+mcol, contraction d split 0-31 / 32-63
  const _Float16* qrow = Qp + (size_t)(q0 + mcol) * HD;
  const v16h qa0 = load_frag16(qrow, lane);
  const v16h qa1 = load_frag16(qrow + 32, lane);

  v8f o[4] = {};
  float mrow[8], lrow[8];
#pragma unroll
  for (int r = 0; r < 8; ++r) { mrow[r] = -1e30f; lrow[r] = 0.f; }

  _Float16* pl = &plds[wv][0];

#pragma unroll 1
  for (int j0 = 0; j0 < SQ; j0 += 32) {
    if (j0 + 32 < SQ)
      __builtin_prefetch(Kp + (size_t)(j0 + 32 + mcol) * HD, 0, 1);

    // ---- scores: two 16x16 tiles (keys j0..j0+15, j0+16..j0+31) ----
    v8f sc[2];
#pragma unroll
    for (int t = 0; t < 2; ++t) {
      const _Float16* kcol = Kp + (size_t)(j0 + t * 16 + mcol) * HD;
      const v16h kb0 = load_frag16(kcol, lane);
      const v16h kb1 = load_frag16(kcol + 32, lane);
      v8f zf = {};
      sc[t] = wmma16(qa0, kb0, zf);
      sc[t] = wmma16(qa1, kb1, sc[t]);
    }

    // ---- online softmax: per-row max/sum via 16-lane xor reductions ----
    float alpha[8];
#pragma unroll
    for (int r = 0; r < 8; ++r) {
      sc[0][r] *= 0.125f;                 // 1/sqrt(64)
      sc[1][r] *= 0.125f;
      float rm = fmaxf(sc[0][r], sc[1][r]);
      rm = fmaxf(rm, __shfl_xor(rm, 1, 32));
      rm = fmaxf(rm, __shfl_xor(rm, 2, 32));
      rm = fmaxf(rm, __shfl_xor(rm, 4, 32));
      rm = fmaxf(rm, __shfl_xor(rm, 8, 32));
      const float mnew = fmaxf(mrow[r], rm);
      alpha[r] = __expf(mrow[r] - mnew);
      const float p0 = __expf(sc[0][r] - mnew);
      const float p1 = __expf(sc[1][r] - mnew);
      sc[0][r] = p0; sc[1][r] = p1;
      float s2 = p0 + p1;
      s2 += __shfl_xor(s2, 1, 32);
      s2 += __shfl_xor(s2, 2, 32);
      s2 += __shfl_xor(s2, 4, 32);
      s2 += __shfl_xor(s2, 8, 32);
      lrow[r] = lrow[r] * alpha[r] + s2;
      mrow[r] = mnew;
    }
#pragma unroll
    for (int t = 0; t < 4; ++t)
#pragma unroll
      for (int r = 0; r < 8; ++r) o[t][r] *= alpha[r];

    // ---- transpose P through LDS: C-layout -> A-fragment layout ----
#pragma unroll
    for (int t = 0; t < 2; ++t)
#pragma unroll
      for (int r = 0; r < 8; ++r)
        pl[(r + mrow_off) * 40 + t * 16 + mcol] = (_Float16)sc[t][r];
    asm volatile("s_wait_dscnt 0" ::: "memory");   // DS is wave-in-order; wait for stores
    const v16h pa = load_frag16(pl + mcol * 40, lane);

    // ---- O += P x V (V transposed => contiguous B-fragment loads) ----
#pragma unroll
    for (int t = 0; t < 4; ++t) {
      const _Float16* vcol = Vp + (size_t)(t * 16 + mcol) * SQ + j0;
      const v16h vb = load_frag16(vcol, lane);
      o[t] = wmma16(pa, vb, o[t]);
    }
  }

  // ---- normalize and store (merge heads: [B,S,H*HD] fp32) ----
  const int bb = bh >> 4, h = bh & 15;
#pragma unroll
  for (int t = 0; t < 4; ++t)
#pragma unroll
    for (int r = 0; r < 8; ++r) {
      const int s = q0 + r + mrow_off;
      out[((size_t)(bb * SQ + s)) * DIM + h * HD + t * 16 + mcol] =
          o[t][r] / lrow[r];
    }
}

// ---------------------------------------------------------------------------
extern "C" void kernel_launch(void* const* d_in, const int* in_sizes, int n_in,
                              void* d_out, int out_size, void* d_ws, size_t ws_size,
                              hipStream_t stream) {
  const float* hs = (const float*)d_in[0];
  const float* Wq = (const float*)d_in[1];
  const float* bq = (const float*)d_in[2];
  const float* Wk = (const float*)d_in[3];
  const float* bk = (const float*)d_in[4];
  const float* Wv = (const float*)d_in[5];
  const float* bv = (const float*)d_in[6];
  float* out = (float*)d_out;

  char* ws = (char*)d_ws;
  _Float16* hsH = (_Float16*)(ws);                                // 16 MiB
  _Float16* wH  = (_Float16*)(ws + (size_t)16 * 1024 * 1024);     //  6 MiB
  _Float16* qH  = (_Float16*)(ws + (size_t)22 * 1024 * 1024);     // 16 MiB
  _Float16* kH  = qH + (size_t)4 * NH * SQ * HD;                  // 16 MiB
  _Float16* vH  = kH + (size_t)4 * NH * SQ * HD;                  // 16 MiB

  const int nHS = 4 * SQ * DIM;      // 8,388,608
  const int nW  = DIM * DIM;         // 1,048,576

  cvt_f32_to_f16<<<nHS / (256 * 8), 256, 0, stream>>>(hs, hsH, nHS);
  cvt_f32_to_f16<<<nW  / (256 * 8), 256, 0, stream>>>(Wq, wH,           nW);
  cvt_f32_to_f16<<<nW  / (256 * 8), 256, 0, stream>>>(Wk, wH + nW,      nW);
  cvt_f32_to_f16<<<nW  / (256 * 8), 256, 0, stream>>>(Wv, wH + 2 * nW,  nW);

  qkv_gemm<<<dim3(64, 8, 3), 128, 0, stream>>>(hsH, wH, bq, bk, bv, qH, kH, vH);
  attn_kernel<<<dim3(SQ / 128, 4 * NH), 256, 0, stream>>>(qH, kH, vH, out);
}
